// Gemma3Attention_15762529976778
// MI455X (gfx1250) — compile-verified, matrix operand order checked by reading, not run on hardware
//
#include <hip/hip_runtime.h>
#include <hip/hip_bf16.h>

// ---------------------------------------------------------------- types
typedef __attribute__((ext_vector_type(16))) __bf16 v16bf;
typedef __attribute__((ext_vector_type(8)))  __bf16 v8bf;
typedef __attribute__((ext_vector_type(4)))  __bf16 v4bf;
typedef __attribute__((ext_vector_type(2)))  __bf16 v2bf;
typedef __attribute__((ext_vector_type(8)))  float  v8f;
typedef __attribute__((ext_vector_type(4)))  float  v4f;

// ---------------------------------------------------------------- dims
constexpr int  Bb    = 4;
constexpr int  QMAXc = 1024;
constexpr int  Dc    = 2560;
constexpr int  Nh    = 8;        // query heads
constexpr int  Kh    = 4;        // kv heads
constexpr int  Hc    = 256;      // head dim
constexpr int  NBc   = 1024;
constexpr int  BSc   = 32;
constexpr int  Mc    = 64;       // blocks per sequence
constexpr int  Tc    = Bb * QMAXc;          // 4096 tokens
constexpr int  NHc   = Nh * Hc;             // 2048
constexpr int  KHc   = Kh * Hc;             // 1024
constexpr int  CTXc  = 1024;
constexpr long long CACHE_PLANE = (long long)NBc * BSc * Kh * Hc; // 33,554,432

// ---------------------------------------------------------------- helpers
__device__ inline __bf16 f2bf(float f) {
  unsigned u = __builtin_bit_cast(unsigned, f);
  unsigned r = u + 0x7FFFu + ((u >> 16) & 1u);
  unsigned short h = (unsigned short)(r >> 16);
  return __builtin_bit_cast(__bf16, h);
}

__device__ inline v8f vzero8() {
  v8f z;
#pragma unroll
  for (int i = 0; i < 8; i++) z[i] = 0.0f;
  return z;
}

// Load a 16-bit WMMA A-layout (or B^T-as-A) fragment: lane<16 takes K{0..7,16..23}
// of row (lane&15), lane>=16 takes K{8..15,24..31}. Two 16-byte loads.
__device__ inline v16bf ldfrag(const __bf16* base, int stride, int row, int koff) {
  const __bf16* p = base + (size_t)row * stride + koff;
  v8bf lo = *(const v8bf*)(p);
  v8bf hi = *(const v8bf*)(p + 16);
  v16bf r;
#pragma unroll
  for (int i = 0; i < 8; i++) { r[i] = lo[i]; r[i + 8] = hi[i]; }
  return r;
}

__device__ inline v8f wmma_bf16(v16bf a, v16bf b, v8f c) {
  return __builtin_amdgcn_wmma_f32_16x16x32_bf16(false, a, false, b, (short)0, c,
                                                 false, false);
}

// Register "pin" tie-points: force all fragments live at one program point so
// the scheduler can't sink each ds_load to just before its WMMA (which would
// serialize every WMMA behind an s_wait_dscnt) and the RA can't alias them.
__device__ inline void pin2(v16bf& a, v16bf& b) {
  asm volatile("" : "+v"(a), "+v"(b));
}
__device__ inline void pin4(v16bf& a, v16bf& b, v16bf& c, v16bf& d) {
  asm volatile("" : "+v"(a), "+v"(b), "+v"(c), "+v"(d));
}
__device__ inline void pin5(v16bf& a, v16bf& b, v16bf& c, v16bf& d, v16bf& e) {
  asm volatile("" : "+v"(a), "+v"(b), "+v"(c), "+v"(d), "+v"(e));
}

// CDNA5 async copy: 16 bytes global -> LDS per lane, tracked by ASYNCcnt.
// LDS operand is the 32-bit local offset (low 32 bits of the generic address,
// per the LDS aperture mapping rules).
__device__ inline void async_ld16(unsigned ldsoff, const __bf16* g) {
  asm volatile("global_load_async_to_lds_b128 %0, %1, off"
               :: "v"(ldsoff), "v"(g)
               : "memory");
}
__device__ inline void wait_async0() {
  asm volatile("s_wait_asynccnt 0x0" ::: "memory");
}

// ---------------------------------------------------------------- fp32 -> bf16
__global__ __launch_bounds__(256) void cvt_f32_bf16(const float* __restrict__ in,
                                                    __bf16* __restrict__ out, int n) {
  int i = (blockIdx.x * 256 + threadIdx.x) * 4;
  if (i < n) {
    v4f v = *(const v4f*)(in + i);
    v4bf o;
#pragma unroll
    for (int j = 0; j < 4; j++) o[j] = f2bf(v[j]);
    *(v4bf*)(out + i) = o;
  }
}

// ---------------------------------------------------------------- GEMM (bf16 x bf16 -> f32)
// C[M,N] = A[M,K] * B[K,N], row-major. Block tile 64x128, BK=32, 256 threads (8 waves).
// Wave w: rows 16*(w>>1), cols 64*(w&1) -> four 16x16 accumulators. Double-buffered
// LDS, one barrier per k-step; A tile staged via async global->LDS copies; B tile
// transposed through registers with paired-k 32-bit LDS stores. All fragments of a
// step are pinned live before the 4 WMMAs so they issue back-to-back after one wait.
__global__ __launch_bounds__(256) void gemm_bf16(const __bf16* __restrict__ A,
                                                 const __bf16* __restrict__ Bm,
                                                 float* __restrict__ C,
                                                 int Mdim, int Ndim, int Kdim) {
  __shared__ __bf16 As[2][64][40];    // row-major A tile (pad 8)
  __shared__ __bf16 Bs[2][128][40];   // transposed B tile: Bs[col][k]
  const int tid  = threadIdx.x;
  const int wave = tid >> 5, lane = tid & 31;
  const int lrow = lane & 15, lko = (lane >> 4) * 8;
  const int rowbase = blockIdx.y * 64;
  const int colbase = blockIdx.x * 128;
  const int wrow = (wave >> 1) * 16;
  const int wcol = (wave & 1) * 64;
  const int arow = tid >> 2, ac0 = (tid & 3) * 8;     // A stage: 64 x 32, 16B/thread
  const int bkk  = (tid >> 4) * 2;                    // B stage: rows kk, kk+1 (even)
  const int bc0  = (tid & 15) * 8;                    // 8 cols per thread

  const unsigned adst0 = (unsigned)(uintptr_t)&As[0][arow][ac0];
  const unsigned adst1 = (unsigned)(uintptr_t)&As[1][arow][ac0];
  const __bf16* aptr = A + (size_t)(rowbase + arow) * Kdim + ac0;
  const __bf16* bptr = Bm + (size_t)bkk * Ndim + colbase + bc0;
  const size_t bstep = (size_t)32 * Ndim;

  v8f acc[4];
#pragma unroll
  for (int c = 0; c < 4; c++) acc[c] = vzero8();

  // ---- prologue: stage k-step 0 into buffer 0
  async_ld16(adst0, aptr);
  {
    v8bf b0 = *(const v8bf*)(bptr);
    v8bf b1 = *(const v8bf*)(bptr + Ndim);
#pragma unroll
    for (int i = 0; i < 8; i++) {
      v2bf p; p[0] = b0[i]; p[1] = b1[i];
      *(v2bf*)(&Bs[0][bc0 + i][bkk]) = p;             // (k, k+1) pair, 4B store
    }
  }
  wait_async0();
  __syncthreads();

  const int steps = Kdim >> 5;
  for (int k = 0; k < steps; k++) {
    const int cur = k & 1, nxt = cur ^ 1;
    if (k + 1 < steps) {
      // stage k+1 into the other buffer (reads of it finished at last barrier)
      async_ld16(nxt ? adst1 : adst0, aptr + (size_t)(k + 1) * 32);
      v8bf b0 = *(const v8bf*)(bptr + (size_t)(k + 1) * bstep);
      v8bf b1 = *(const v8bf*)(bptr + (size_t)(k + 1) * bstep + Ndim);
#pragma unroll
      for (int i = 0; i < 8; i++) {
        v2bf p; p[0] = b0[i]; p[1] = b1[i];
        *(v2bf*)(&Bs[nxt][bc0 + i][bkk]) = p;
      }
    }
    // compute on current buffer: load + pin all fragments, then 4 WMMAs back-to-back
    v16bf a  = ldfrag(&As[cur][0][0], 40, wrow + lrow, lko);
    v16bf f0 = ldfrag(&Bs[cur][0][0], 40, wcol + 0  + lrow, lko);
    v16bf f1 = ldfrag(&Bs[cur][0][0], 40, wcol + 16 + lrow, lko);
    v16bf f2 = ldfrag(&Bs[cur][0][0], 40, wcol + 32 + lrow, lko);
    v16bf f3 = ldfrag(&Bs[cur][0][0], 40, wcol + 48 + lrow, lko);
    pin5(a, f0, f1, f2, f3);
    acc[0] = wmma_bf16(a, f0, acc[0]);
    acc[1] = wmma_bf16(a, f1, acc[1]);
    acc[2] = wmma_bf16(a, f2, acc[2]);
    acc[3] = wmma_bf16(a, f3, acc[3]);

    wait_async0();
    __syncthreads();
  }

  const int crow = rowbase + wrow + 8 * (lane >> 4);
  const int ccol = colbase + wcol + lrow;
#pragma unroll
  for (int c = 0; c < 4; c++)
#pragma unroll
    for (int j = 0; j < 8; j++)
      C[(size_t)(crow + j) * Ndim + ccol + c * 16] = acc[c][j];
}

// ---------------------------------------------------------------- rmsnorm + rope (one wave / row of 256)
__device__ inline void norm_rope8(const float* __restrict__ x,
                                  const float* __restrict__ gamma,
                                  float pos, int lane, float scale,
                                  float* __restrict__ out) {
  v4f a = *(const v4f*)(x + lane * 8);
  v4f b = *(const v4f*)(x + lane * 8 + 4);
  float v[8];
#pragma unroll
  for (int i = 0; i < 4; i++) { v[i] = a[i]; v[4 + i] = b[i]; }
  float ss = 0.0f;
#pragma unroll
  for (int i = 0; i < 8; i++) ss += v[i] * v[i];
#pragma unroll
  for (int m = 1; m < 32; m <<= 1) ss += __shfl_xor(ss, m, 32);
  float rinv = rsqrtf(ss * (1.0f / 256.0f) + 1e-6f);
#pragma unroll
  for (int i = 0; i < 8; i++) {
    int h = lane * 8 + i;
    float y = (gamma[h] + 1.0f) * v[i] * rinv;
    float other = __shfl_xor(y, 16, 32);        // partner h +/- 128
    int hh = h & 127;
    float ang = pos * __powf(1.0e6f, -(float)hh * (1.0f / 128.0f));
    float cs = __cosf(ang), sn = __sinf(ang);
    out[i] = (y * cs + ((lane < 16) ? -other : other) * sn) * scale;
  }
}

__global__ __launch_bounds__(256) void finalize_q(const float* __restrict__ qraw,
                                                  const float* __restrict__ gamma,
                                                  const int* __restrict__ pos,
                                                  __bf16* __restrict__ qb) {
  int row  = blockIdx.x * 8 + (threadIdx.x >> 5);   // [0, T*N)
  int lane = threadIdx.x & 31;
  int t = row >> 3;
  float o[8];
  norm_rope8(qraw + (size_t)row * 256, gamma, (float)pos[t], lane, 0.0625f, o);
  v4bf p0, p1;
#pragma unroll
  for (int i = 0; i < 4; i++) { p0[i] = f2bf(o[i]); p1[i] = f2bf(o[4 + i]); }
  __bf16* dst = qb + (size_t)row * 256 + lane * 8;
  *(v4bf*)(dst)     = p0;
  *(v4bf*)(dst + 4) = p1;
}

__global__ __launch_bounds__(256) void finalize_kv(const float* __restrict__ kraw,
                                                   const float* __restrict__ vraw,
                                                   const float* __restrict__ gamma,
                                                   const int* __restrict__ pos,
                                                   const int* __restrict__ btab,
                                                   float* __restrict__ cache) {
  int row  = blockIdx.x * 8 + (threadIdx.x >> 5);   // [0, T*K)
  int lane = threadIdx.x & 31;
  int t  = row >> 2;
  int kh = row & 3;
  int p  = pos[t];
  int b  = t >> 10;                                  // QMAX = 1024
  int blk  = btab[b * Mc + (p >> 5)];
  int slot = p & 31;
  size_t base = ((size_t)(blk * BSc + slot) * Kh + kh) * Hc;

  float o[8];
  norm_rope8(kraw + (size_t)row * 256, gamma, (float)p, lane, 1.0f, o);
  float* kd = cache + base + lane * 8;
  v4f k0; v4f k1;
#pragma unroll
  for (int i = 0; i < 4; i++) { k0[i] = o[i]; k1[i] = o[4 + i]; }
  *(v4f*)(kd)     = k0;
  *(v4f*)(kd + 4) = k1;

  const float* vr = vraw + (size_t)row * 256 + lane * 8;
  float* vd = cache + CACHE_PLANE + base + lane * 8;
  *(v4f*)(vd)     = *(const v4f*)(vr);
  *(v4f*)(vd + 4) = *(const v4f*)(vr + 4);
}

// ---------------------------------------------------------------- flash attention
// grid: B * N * (QMAX/128).  Each wave owns 16 q rows; S-tiles of 32 keys.
__global__ __launch_bounds__(256) void attention(const __bf16* __restrict__ qb,
                                                 const float* __restrict__ cache,
                                                 const int* __restrict__ btab,
                                                 __bf16* __restrict__ ctxb) {
  __shared__ __bf16 Kt[32][264];      // [key][h]  (bf16, pad 8)
  __shared__ __bf16 Vt[256][40];      // [h][key]  transposed V
  __shared__ __bf16 Pb[8][16][40];    // per-wave probs, A-layout staging

  const int qt = blockIdx.x & 7;
  const int n  = (blockIdx.x >> 3) & 7;
  const int b  = blockIdx.x >> 6;
  const int kh = n >> 1;                               // G = N/K = 2
  const int wave = threadIdx.x >> 5, lane = threadIdx.x & 31;
  const int lrow = lane & 15, lko = (lane >> 4) * 8;

  // Q fragments resident in VGPRs (16 rows x 256, bf16)
  const int trow = b * QMAXc + qt * 128 + wave * 16 + lrow;
  const __bf16* qptr = qb + (size_t)trow * NHc + n * Hc;
  v16bf aq[8];
#pragma unroll
  for (int c = 0; c < 8; c++) aq[c] = ldfrag(qptr, 0, 0, c * 32 + lko);

  v8f o[16];
#pragma unroll
  for (int i = 0; i < 16; i++) o[i] = vzero8();
  float rm[8], rl[8];
#pragma unroll
  for (int j = 0; j < 8; j++) { rm[j] = -3.0e38f; rl[j] = 0.0f; }

  const int half = lane >> 4;
  const int qrow0 = qt * 128 + wave * 16 + 8 * half;   // + j gives this lane's row
  const int smax = min(2048, CTXc + qt * 128 + 128);

  const float* kplane = cache;
  const float* vplane = cache + CACHE_PLANE;
  const int skey = (threadIdx.x >> 4) * 2;             // staging: keys skey, skey+1
  const int sh0  = (threadIdx.x & 15) * 16;            // staging: 16-h chunk

  for (int s0 = 0; s0 < smax; s0 += 32) {
    // ---- stage K and V^T tiles (f32 -> bf16); thread covers 2 keys x 16 h
    int blk  = btab[b * Mc + ((s0 + skey) >> 5)];
    int slot = (s0 + skey) & 31;                       // slot+1 = second key
    size_t rbase = ((size_t)(blk * BSc + slot) * Kh + kh) * Hc;
    v4f ka[4], kb2[4], va[4], vb2[4];
#pragma unroll
    for (int i = 0; i < 4; i++) {
      ka[i]  = *(const v4f*)(kplane + rbase + sh0 + i * 4);
      kb2[i] = *(const v4f*)(kplane + rbase + (size_t)Kh * Hc + sh0 + i * 4);
      va[i]  = *(const v4f*)(vplane + rbase + sh0 + i * 4);
      vb2[i] = *(const v4f*)(vplane + rbase + (size_t)Kh * Hc + sh0 + i * 4);
    }
    // prefetch next tile's rows while this tile computes
    if (s0 + 32 < smax) {
      int nblk  = btab[b * Mc + ((s0 + 32 + skey) >> 5)];
      size_t nb = ((size_t)(nblk * BSc + slot) * Kh + kh) * Hc;
      __builtin_prefetch(kplane + nb + sh0, 0, 0);
      __builtin_prefetch(vplane + nb + sh0, 0, 0);
    }
    __syncthreads();                                   // prior tile consumed
#pragma unroll
    for (int i = 0; i < 4; i++) {
      v4bf c0, c1;
#pragma unroll
      for (int j = 0; j < 4; j++) { c0[j] = f2bf(ka[i][j]); c1[j] = f2bf(kb2[i][j]); }
      *(v4bf*)(&Kt[skey][sh0 + i * 4])     = c0;       // 8B stores
      *(v4bf*)(&Kt[skey + 1][sh0 + i * 4]) = c1;
#pragma unroll
      for (int j = 0; j < 4; j++) {
        v2bf p; p[0] = f2bf(va[i][j]); p[1] = f2bf(vb2[i][j]);
        *(v2bf*)(&Vt[sh0 + i * 4 + j][skey]) = p;      // paired-key 4B stores
      }
    }
    __syncthreads();

    // ---- scores: 16q x 32s, K-dim 256 in 8 WMMA steps.
    // Next-chunk K fragments load while current WMMAs run; the pin after the
    // WMMAs keeps the s_wait_dscnt behind them.
    v8f sa = vzero8(), sb = vzero8();
    v16bf kb0 = ldfrag(&Kt[0][0], 264, lrow, lko);
    v16bf kb1 = ldfrag(&Kt[0][0], 264, 16 + lrow, lko);
#pragma unroll
    for (int c = 0; c < 8; c++) {
      v16bf n0 = kb0, n1 = kb1;
      if (c < 7) {
        n0 = ldfrag(&Kt[0][0], 264, lrow, (c + 1) * 32 + lko);
        n1 = ldfrag(&Kt[0][0], 264, 16 + lrow, (c + 1) * 32 + lko);
      }
      sa = wmma_bf16(aq[c], kb0, sa);
      sb = wmma_bf16(aq[c], kb1, sb);
      pin2(n0, n1);
      kb0 = n0; kb1 = n1;
    }

    // ---- causal mask + online softmax
    const int scol0 = s0 + lrow;
    const int scol1 = scol0 + 16;
    float nm[8];
#pragma unroll
    for (int j = 0; j < 8; j++) {
      int qpos = CTXc + qrow0 + j;
      if (scol0 > qpos) sa[j] = -3.0e38f;
      if (scol1 > qpos) sb[j] = -3.0e38f;
      nm[j] = fmaxf(sa[j], sb[j]);
    }
#pragma unroll
    for (int m = 1; m < 16; m <<= 1)
#pragma unroll
      for (int j = 0; j < 8; j++) nm[j] = fmaxf(nm[j], __shfl_xor(nm[j], m, 32));
    float sc[8], rs[8];
#pragma unroll
    for (int j = 0; j < 8; j++) {
      nm[j] = fmaxf(nm[j], rm[j]);
      sc[j] = __expf(rm[j] - nm[j]);
      rm[j] = nm[j];
      sa[j] = __expf(sa[j] - nm[j]);
      sb[j] = __expf(sb[j] - nm[j]);
      rs[j] = sa[j] + sb[j];
    }
#pragma unroll
    for (int m = 1; m < 16; m <<= 1)
#pragma unroll
      for (int j = 0; j < 8; j++) rs[j] += __shfl_xor(rs[j], m, 32);
#pragma unroll
    for (int j = 0; j < 8; j++) rl[j] = rl[j] * sc[j] + rs[j];
#pragma unroll
    for (int ht = 0; ht < 16; ht++)
#pragma unroll
      for (int j = 0; j < 8; j++) o[ht][j] *= sc[j];

    // ---- probs C-layout -> A-layout via per-wave LDS buffer
#pragma unroll
    for (int j = 0; j < 8; j++) {
      int m = j + 8 * half;
      Pb[wave][m][lrow]      = f2bf(sa[j]);
      Pb[wave][m][lrow + 16] = f2bf(sb[j]);
    }
    v16bf ap = ldfrag(&Pb[wave][0][0], 40, lrow, lko);

    // ---- O += P(16x32) * V(32x256): 16 WMMAs, groups of 4, next group's V
    // fragments prefetched behind the current group's WMMAs.
    v16bf bv0 = ldfrag(&Vt[0][0], 40, 0 * 16 + lrow, lko);
    v16bf bv1 = ldfrag(&Vt[0][0], 40, 1 * 16 + lrow, lko);
    v16bf bv2 = ldfrag(&Vt[0][0], 40, 2 * 16 + lrow, lko);
    v16bf bv3 = ldfrag(&Vt[0][0], 40, 3 * 16 + lrow, lko);
#pragma unroll
    for (int g = 0; g < 4; g++) {
      v16bf n0 = bv0, n1 = bv1, n2 = bv2, n3 = bv3;
      if (g < 3) {
        n0 = ldfrag(&Vt[0][0], 40, (g * 4 + 4) * 16 + lrow, lko);
        n1 = ldfrag(&Vt[0][0], 40, (g * 4 + 5) * 16 + lrow, lko);
        n2 = ldfrag(&Vt[0][0], 40, (g * 4 + 6) * 16 + lrow, lko);
        n3 = ldfrag(&Vt[0][0], 40, (g * 4 + 7) * 16 + lrow, lko);
      }
      o[g * 4 + 0] = wmma_bf16(ap, bv0, o[g * 4 + 0]);
      o[g * 4 + 1] = wmma_bf16(ap, bv1, o[g * 4 + 1]);
      o[g * 4 + 2] = wmma_bf16(ap, bv2, o[g * 4 + 2]);
      o[g * 4 + 3] = wmma_bf16(ap, bv3, o[g * 4 + 3]);
      pin4(n0, n1, n2, n3);
      bv0 = n0; bv1 = n1; bv2 = n2; bv3 = n3;
    }
  }

  // ---- normalize and store ctx (bf16)
  float rli[8];
#pragma unroll
  for (int j = 0; j < 8; j++) rli[j] = 1.0f / rl[j];
  const int tg = b * QMAXc + qt * 128 + wave * 16;
#pragma unroll
  for (int ht = 0; ht < 16; ht++)
#pragma unroll
    for (int j = 0; j < 8; j++) {
      int m = j + 8 * half;
      ctxb[(size_t)(tg + m) * NHc + n * Hc + ht * 16 + lrow] = f2bf(o[ht][j] * rli[j]);
    }
}

// ---------------------------------------------------------------- launch
extern "C" void kernel_launch(void* const* d_in, const int* in_sizes, int n_in,
                              void* d_out, int out_size, void* d_ws, size_t ws_size,
                              hipStream_t stream) {
  const float* x      = (const float*)d_in[0];
  const float* kv     = (const float*)d_in[1];
  const float* Wq     = (const float*)d_in[2];
  const float* Wk     = (const float*)d_in[3];
  const float* Wv     = (const float*)d_in[4];
  const float* Wo     = (const float*)d_in[5];
  const float* qgam   = (const float*)d_in[6];
  const float* kgam   = (const float*)d_in[7];
  const int*   pos    = (const int*)d_in[8];
  const int*   btab   = (const int*)d_in[10];

  float* out_cache = (float*)d_out;
  float* out_o     = out_cache + 2 * CACHE_PLANE;

  // workspace carve-up (bytes)
  char* w = (char*)d_ws;
  __bf16* xb   = (__bf16*)(w);                          // T*D
  __bf16* wqb  = (__bf16*)(w + 20971520);               // D*NH
  __bf16* wkb  = (__bf16*)(w + 31457280);               // D*KH
  __bf16* wvb  = (__bf16*)(w + 36700160);               // D*KH
  __bf16* wob  = (__bf16*)(w + 41943040);               // NH*D
  float*  qraw = (float*) (w + 52428800);               // T*NH f32
  float*  kraw = (float*) (w + 85983232);               // T*KH f32
  float*  vraw = (float*) (w + 102760448);              // T*KH f32
  __bf16* qbb  = (__bf16*)(w + 119537664);              // T*NH bf16
  __bf16* ctxb = (__bf16*)(w + 136314880);              // T*NH bf16

  // 1) carry the kv cache through to the output (then scatter updates into it)
  hipMemcpyAsync(out_cache, kv, (size_t)2 * CACHE_PLANE * sizeof(float),
                 hipMemcpyDeviceToDevice, stream);

  // 2) bf16 conversions
  cvt_f32_bf16<<<(Tc * Dc) / 1024, 256, 0, stream>>>(x, xb, Tc * Dc);
  cvt_f32_bf16<<<(Dc * NHc) / 1024, 256, 0, stream>>>(Wq, wqb, Dc * NHc);
  cvt_f32_bf16<<<(Dc * KHc) / 1024, 256, 0, stream>>>(Wk, wkb, Dc * KHc);
  cvt_f32_bf16<<<(Dc * KHc) / 1024, 256, 0, stream>>>(Wv, wvb, Dc * KHc);
  cvt_f32_bf16<<<(NHc * Dc) / 1024, 256, 0, stream>>>(Wo, wob, NHc * Dc);

  // 3) QKV projections
  gemm_bf16<<<dim3(NHc / 128, Tc / 64), 256, 0, stream>>>(xb, wqb, qraw, Tc, NHc, Dc);
  gemm_bf16<<<dim3(KHc / 128, Tc / 64), 256, 0, stream>>>(xb, wkb, kraw, Tc, KHc, Dc);
  gemm_bf16<<<dim3(KHc / 128, Tc / 64), 256, 0, stream>>>(xb, wvb, vraw, Tc, KHc, Dc);

  // 4) rmsnorm + rope; scatter k/v into output cache
  finalize_q<<<(Tc * Nh) / 8, 256, 0, stream>>>(qraw, qgam, pos, qbb);
  finalize_kv<<<(Tc * Kh) / 8, 256, 0, stream>>>(kraw, vraw, kgam, pos, btab, out_cache);

  // 5) flash attention over the updated cache
  attention<<<Bb * Nh * (QMAXc / 128), 256, 0, stream>>>(qbb, out_cache, btab, ctxb);

  // 6) output projection
  gemm_bf16<<<dim3(Dc / 128, Tc / 64), 256, 0, stream>>>(ctxb, wob, out_o, Tc, Dc, NHc);
}
